// TRANSFOMER_ENCODER_28784870818154
// MI455X (gfx1250) — matrix operand order repaired and checked by reading, not compiled, not run on hardware
//
#include <hip/hip_runtime.h>
#include <hip/hip_bf16.h>
#include <math.h>

// ---------------------------------------------------------------------------
// CDNA5 (gfx1250) transformer encoder block.
// All GEMMs run on v_wmma_f32_16x16x32_bf16 (bf16 operands, fp32 accum),
// converting fp32 -> bf16 during the global->LDS staging pass.
// Block tile 128x128x32, 256 threads = 8 waves, each wave = 32x64 strip
// = 8 WMMA issues per K-step. Global loads for step k+1 are issued into a
// persistent register buffer while step k's WMMAs execute (software pipeline).
// ---------------------------------------------------------------------------

typedef __attribute__((ext_vector_type(16))) __bf16 v16bf;
typedef __attribute__((ext_vector_type(8)))  __bf16 v8bf;
typedef __attribute__((ext_vector_type(4)))  __bf16 v4bf;
typedef __attribute__((ext_vector_type(8)))  float  v8f;

#define BM 128
#define BN 128
#define BKK 32
#define ASTR 48   // bf16 elements per LDS row (32 data + 16 pad); 96B rows, 32B-aligned
#define BSTR 48

enum { EPI_NONE = 0, EPI_GELU = 1 };

__device__ __forceinline__ float gelu_erf(float x) {
    // exact-erf GELU (matches torch nn.GELU default / jax approximate=False)
    return 0.5f * x * (1.0f + erff(x * 0.7071067811865475f));
}

// C[M,N] = epi(alpha * A[M,K] @ op(B) + bias),  op(B)=B[K,N] or B^T with B[N,K]
// Requires: M%128==0, N%128==0, K%32==0 (true for all shapes in this model).
template <int TRANSB, int EPI>
__global__ __launch_bounds__(256)
void gemm_wmma_bf16(const float* __restrict__ A,
                    const float* __restrict__ B,
                    const float* __restrict__ bias,
                    float* __restrict__ C,
                    int M, int N, int K, float alpha)
{
    __shared__ __bf16 As[BM * ASTR];   // [m][k], k-contiguous
    __shared__ __bf16 Bt[BN * BSTR];   // [n][k], k-contiguous (transposed tile)

    const int tid  = threadIdx.x;
    const int lane = tid & 31;
    const int wid  = tid >> 5;       // 8 waves
    const int hlf  = lane >> 4;      // lane half (wave32 WMMA layout)
    const int l16  = lane & 15;
    const int wm   = wid & 3;        // wave row-slice: 4 x 32 rows
    const int wn   = wid >> 2;       // wave col-slice: 2 x 64 cols

    const int bm0 = blockIdx.y * BM;
    const int bn0 = blockIdx.x * BN;

    v8f acc[2][4] = {};

    // persistent register stage buffers (software pipeline across K-steps)
    float4 areg[4], breg[4];

    auto load_tiles = [&](int k0) {
        #pragma unroll
        for (int i = 0; i < 4; ++i) {
            const int t2  = tid + i * 256;      // 0..1023
            const int row = t2 >> 3;            // 0..127
            const int c4  = t2 & 7;
            areg[i] = *reinterpret_cast<const float4*>(
                A + (size_t)(bm0 + row) * K + k0 + c4 * 4);
        }
        if (TRANSB) {
            #pragma unroll
            for (int i = 0; i < 4; ++i) {
                const int t2 = tid + i * 256;
                const int n  = t2 >> 3;
                const int c4 = t2 & 7;
                breg[i] = *reinterpret_cast<const float4*>(
                    B + (size_t)(bn0 + n) * K + k0 + c4 * 4);
            }
        } else {
            #pragma unroll
            for (int i = 0; i < 4; ++i) {
                const int t2 = tid + i * 256;
                const int k  = t2 >> 5;          // 0..31
                const int n4 = t2 & 31;          // group of 4 cols (128 cols)
                breg[i] = *reinterpret_cast<const float4*>(
                    B + (size_t)(k0 + k) * N + bn0 + n4 * 4);
            }
        }
    };

    auto stage_to_lds = [&]() {
        #pragma unroll
        for (int i = 0; i < 4; ++i) {
            const int t2  = tid + i * 256;
            const int row = t2 >> 3;
            const int c4  = t2 & 7;
            v4bf w;
            w[0] = (__bf16)areg[i].x; w[1] = (__bf16)areg[i].y;
            w[2] = (__bf16)areg[i].z; w[3] = (__bf16)areg[i].w;
            *reinterpret_cast<v4bf*>(&As[row * ASTR + c4 * 4]) = w;
        }
        if (TRANSB) {
            #pragma unroll
            for (int i = 0; i < 4; ++i) {
                const int t2 = tid + i * 256;
                const int n  = t2 >> 3;
                const int c4 = t2 & 7;
                v4bf w;
                w[0] = (__bf16)breg[i].x; w[1] = (__bf16)breg[i].y;
                w[2] = (__bf16)breg[i].z; w[3] = (__bf16)breg[i].w;
                *reinterpret_cast<v4bf*>(&Bt[n * BSTR + c4 * 4]) = w;
            }
        } else {
            #pragma unroll
            for (int i = 0; i < 4; ++i) {
                const int t2 = tid + i * 256;
                const int k  = t2 >> 5;
                const int n4 = t2 & 31;
                Bt[(n4 * 4 + 0) * BSTR + k] = (__bf16)breg[i].x;
                Bt[(n4 * 4 + 1) * BSTR + k] = (__bf16)breg[i].y;
                Bt[(n4 * 4 + 2) * BSTR + k] = (__bf16)breg[i].z;
                Bt[(n4 * 4 + 3) * BSTR + k] = (__bf16)breg[i].w;
            }
        }
    };

    load_tiles(0);   // prologue: first tile in flight

    for (int k0 = 0; k0 < K; k0 += BKK) {
        stage_to_lds();          // waits on in-flight loads, cvt -> LDS
        __syncthreads();

        if (k0 + BKK < K) load_tiles(k0 + BKK);   // next tile in flight, no wait

        // ---- A fragments (16x32 bf16, contiguous k order, k = 16*hlf + s):
        //   lane<16 : K 0-15 ; lane>=16 : K 16-31
        v16bf afrag[2];
        #pragma unroll
        for (int sm = 0; sm < 2; ++sm) {
            const __bf16* arow = &As[(wm * 32 + sm * 16 + l16) * ASTR];
            const v8bf alo = *reinterpret_cast<const v8bf*>(arow + hlf * 16);
            const v8bf ahi = *reinterpret_cast<const v8bf*>(arow + hlf * 16 + 8);
            afrag[sm] = __builtin_shufflevector(
                alo, ahi, 0, 1, 2, 3, 4, 5, 6, 7, 8, 9, 10, 11, 12, 13, 14, 15);
        }
        // ---- B fragments (32x16): lane holds column n=l16, K half per lane half
        v16bf bfrag[4];
        #pragma unroll
        for (int sn = 0; sn < 4; ++sn) {
            bfrag[sn] = *reinterpret_cast<const v16bf*>(
                &Bt[(wn * 64 + sn * 16 + l16) * BSTR + hlf * 16]);
        }

        #pragma unroll
        for (int sm = 0; sm < 2; ++sm) {
            #pragma unroll
            for (int sn = 0; sn < 4; ++sn) {
                acc[sm][sn] = __builtin_amdgcn_wmma_f32_16x16x32_bf16(
                    false, afrag[sm], false, bfrag[sn], (short)0,
                    acc[sm][sn], false, false);
            }
        }

        __syncthreads();
    }

    // ---- epilogue: documented C/D layout (VGPR r -> M = hlf*8 + r, lane -> N)
    #pragma unroll
    for (int sm = 0; sm < 2; ++sm) {
        const int row0 = bm0 + wm * 32 + sm * 16 + hlf * 8;
        #pragma unroll
        for (int sn = 0; sn < 4; ++sn) {
            const int col = bn0 + wn * 64 + sn * 16 + l16;
            const float bb = bias ? bias[col] : 0.0f;
            #pragma unroll
            for (int r = 0; r < 8; ++r) {
                float v = acc[sm][sn][r] * alpha + bb;
                if (EPI == EPI_GELU) v = gelu_erf(v);
                C[(size_t)(row0 + r) * N + col] = v;
            }
        }
    }
}

// ---------------------------------------------------------------------------
// Row softmax, in place. One block (256 threads = 8 waves) per row.
// ---------------------------------------------------------------------------
__global__ __launch_bounds__(256)
void softmax_rows(float* __restrict__ S, int cols)
{
    const int row = blockIdx.x;
    const size_t base = (size_t)row * cols;
    const int tid = threadIdx.x;
    const int wid = tid >> 5, lane = tid & 31;
    __shared__ float red[8];

    float m = -3.402823466e38f;
    for (int c = tid; c < cols; c += 256) m = fmaxf(m, S[base + c]);
    #pragma unroll
    for (int off = 16; off > 0; off >>= 1) m = fmaxf(m, __shfl_xor(m, off, 32));
    if (lane == 0) red[wid] = m;
    __syncthreads();
    m = red[0];
    #pragma unroll
    for (int i = 1; i < 8; ++i) m = fmaxf(m, red[i]);
    __syncthreads();

    float sum = 0.0f;
    for (int c = tid; c < cols; c += 256) {
        const float e = __expf(S[base + c] - m);
        S[base + c] = e;
        sum += e;
    }
    #pragma unroll
    for (int off = 16; off > 0; off >>= 1) sum += __shfl_xor(sum, off, 32);
    if (lane == 0) red[wid] = sum;
    __syncthreads();
    sum = 0.0f;
    #pragma unroll
    for (int i = 0; i < 8; ++i) sum += red[i];
    const float inv = 1.0f / sum;
    for (int c = tid; c < cols; c += 256) S[base + c] *= inv;
}

// ---------------------------------------------------------------------------
// out = LayerNorm(y + res) * g + b.  One block per row, D == 4096.
// ---------------------------------------------------------------------------
__global__ __launch_bounds__(256)
void residual_layernorm(const float* __restrict__ y,
                        const float* __restrict__ res,
                        const float* __restrict__ g,
                        const float* __restrict__ b,
                        float* __restrict__ out, int D)
{
    const int row = blockIdx.x;
    const size_t base = (size_t)row * D;
    const int tid = threadIdx.x;
    const int wid = tid >> 5, lane = tid & 31;

    float vloc[16];
    float s = 0.0f, s2 = 0.0f;
    #pragma unroll
    for (int i = 0; i < 16; ++i) {
        const int c = tid + i * 256;
        const float v = y[base + c] + res[base + c];
        vloc[i] = v;
        s += v; s2 += v * v;
    }
    #pragma unroll
    for (int off = 16; off > 0; off >>= 1) {
        s  += __shfl_xor(s, off, 32);
        s2 += __shfl_xor(s2, off, 32);
    }
    __shared__ float rs[8], rs2[8];
    if (lane == 0) { rs[wid] = s; rs2[wid] = s2; }
    __syncthreads();
    s = 0.0f; s2 = 0.0f;
    #pragma unroll
    for (int i = 0; i < 8; ++i) { s += rs[i]; s2 += rs2[i]; }

    const float invD = 1.0f / (float)D;
    const float mu   = s * invD;
    const float var  = s2 * invD - mu * mu;
    const float rstd = rsqrtf(var + 1e-5f);
    #pragma unroll
    for (int i = 0; i < 16; ++i) {
        const int c = tid + i * 256;
        out[base + c] = (vloc[i] - mu) * rstd * g[c] + b[c];
    }
}

// ---------------------------------------------------------------------------
// Host-side orchestration
// ---------------------------------------------------------------------------
extern "C" void kernel_launch(void* const* d_in, const int* in_sizes, int n_in,
                              void* d_out, int out_size, void* d_ws, size_t ws_size,
                              hipStream_t stream)
{
    (void)in_sizes; (void)n_in; (void)out_size; (void)ws_size;

    const int Bb = 2, Ss = 2048, Dd = 4096, Hd = 4 * Dd;
    const int NT = Bb * Ss;                 // 4096 tokens

    const float* x   = (const float*)d_in[0];
    const float* Wq  = (const float*)d_in[1];
    const float* bq  = (const float*)d_in[2];
    const float* Wk  = (const float*)d_in[3];
    const float* bk  = (const float*)d_in[4];
    const float* Wv  = (const float*)d_in[5];
    const float* bv  = (const float*)d_in[6];
    const float* Wo  = (const float*)d_in[7];
    const float* bo  = (const float*)d_in[8];
    const float* g1  = (const float*)d_in[9];
    const float* b1  = (const float*)d_in[10];
    const float* g2  = (const float*)d_in[11];
    const float* b2  = (const float*)d_in[12];
    const float* W1  = (const float*)d_in[13];
    const float* bi1 = (const float*)d_in[14];
    const float* W2  = (const float*)d_in[15];
    const float* bi2 = (const float*)d_in[16];

    float* ws = (float*)d_ws;
    const size_t tokD = (size_t)NT * Dd;        // 16.78M floats
    const size_t ssz  = (size_t)Bb * Ss * Ss;   // 8.39M floats

    float* Q   = ws;
    float* Kp  = Q  + tokD;
    float* V   = Kp + tokD;
    float* Sc  = V  + tokD;
    float* O   = Sc + ssz;
    float* X1  = O  + tokD;
    float* Hh  = X1 + tokD;                     // NT x 16384
    float* ATT = Q;                             // Q dead after scores
    float* Y2  = O;                             // O dead after LN1

    const dim3 blk(256);
    auto grd = [](int n, int m) { return dim3(n / BN, m / BM); };

    // QKV projections
    gemm_wmma_bf16<0, EPI_NONE><<<grd(Dd, NT), blk, 0, stream>>>(x, Wq, bq, Q,  NT, Dd, Dd, 1.0f);
    gemm_wmma_bf16<0, EPI_NONE><<<grd(Dd, NT), blk, 0, stream>>>(x, Wk, bk, Kp, NT, Dd, Dd, 1.0f);
    gemm_wmma_bf16<0, EPI_NONE><<<grd(Dd, NT), blk, 0, stream>>>(x, Wv, bv, V,  NT, Dd, Dd, 1.0f);

    // scores = Q @ K^T / sqrt(D)   (per batch)
    for (int b = 0; b < Bb; ++b) {
        gemm_wmma_bf16<1, EPI_NONE><<<grd(Ss, Ss), blk, 0, stream>>>(
            Q + (size_t)b * Ss * Dd, Kp + (size_t)b * Ss * Dd, nullptr,
            Sc + (size_t)b * Ss * Ss, Ss, Ss, Dd, 0.015625f /* 1/64 */);
    }
    softmax_rows<<<dim3(Bb * Ss), blk, 0, stream>>>(Sc, Ss);

    // attn = P @ V   (per batch; output reuses Q buffer)
    for (int b = 0; b < Bb; ++b) {
        gemm_wmma_bf16<0, EPI_NONE><<<grd(Dd, Ss), blk, 0, stream>>>(
            Sc + (size_t)b * Ss * Ss, V + (size_t)b * Ss * Dd, nullptr,
            ATT + (size_t)b * Ss * Dd, Ss, Dd, Ss, 1.0f);
    }

    // O-projection + residual + LN1
    gemm_wmma_bf16<0, EPI_NONE><<<grd(Dd, NT), blk, 0, stream>>>(ATT, Wo, bo, O, NT, Dd, Dd, 1.0f);
    residual_layernorm<<<dim3(NT), blk, 0, stream>>>(O, x, g1, b1, X1, Dd);

    // MLP with fused exact-erf GELU
    gemm_wmma_bf16<0, EPI_GELU><<<grd(Hd, NT), blk, 0, stream>>>(X1, W1, bi1, Hh, NT, Hd, Dd, 1.0f);
    gemm_wmma_bf16<0, EPI_NONE><<<grd(Dd, NT), blk, 0, stream>>>(Hh, W2, bi2, Y2, NT, Dd, Hd, 1.0f);

    // residual + LN2 -> output
    residual_layernorm<<<dim3(NT), blk, 0, stream>>>(Y2, X1, g2, b2, (float*)d_out, Dd);
}
